// FlowLevel_8151847928430
// MI455X (gfx1250) — matrix operand order repaired and checked by reading, not compiled, not run on hardware
//
#include <hip/hip_runtime.h>

// ---------------------------------------------------------------------------
// DICL-style flow level for MI455X (gfx1250, wave32, WMMA).
// All large convs (incl. the transposed conv, via zero-stuffed input) run as
// implicit GEMM on v_wmma_f32_16x16x32_f16:
//   * weights pre-packed zero-padded [Cout16 x K64] f16 -> A fragment = two
//     unconditional 16B vector loads (fragment K-runs are contiguous)
//   * im2col B tile staged per block into LDS in fragment order with
//     division-free incremental (ci,ky,kx) walking; 64 K-rows per barrier
//     pair -> 2 WMMAs per staging round, B fragment = contiguous 32B LDS read
// ---------------------------------------------------------------------------

typedef _Float16 f16;
typedef __attribute__((ext_vector_type(16))) _Float16 v16h;
typedef __attribute__((ext_vector_type(8)))  _Float16 v8h;
typedef __attribute__((ext_vector_type(8)))  float    v8f;

#define HW_FULL (48 * 64)
#define NB 162  // b * 9 * 9

// ---------------------------------------------------------------------------
// Weight repack: f32 [Cout][Cin*taps] -> f16 [ceil16(Cout)][ceil64(Cin*taps)]
// ---------------------------------------------------------------------------
__global__ void pack_wgt_kernel(const float* __restrict__ w, f16* __restrict__ o,
                                int Cout, int Ktot, int Kpad, int rows) {
  int i = blockIdx.x * blockDim.x + threadIdx.x;
  if (i >= rows * Kpad) return;
  int r = i / Kpad, k = i - r * Kpad;
  o[i] = (r < Cout && k < Ktot) ? (f16)w[r * Ktot + k] : (f16)0;
}

// Deconv weights: m4_w [cin=64][cout=32][4][4] -> flipped/transposed
// conv weights o[co][ci*16 + ky*4 + kx] = w[ci][co][3-ky][3-kx], K=1024.
__global__ void pack_deconv_kernel(const float* __restrict__ w, f16* __restrict__ o) {
  int i = blockIdx.x * blockDim.x + threadIdx.x;
  if (i >= 32 * 1024) return;
  int co = i >> 10, k = i & 1023;
  int ci = k >> 4, r = k & 15;
  int ky = r >> 2, kx = r & 3;
  o[i] = (f16)w[((ci * 32 + co) * 4 + (3 - ky)) * 4 + (3 - kx)];
}

// ---------------------------------------------------------------------------
// Cost-volume assembly: mvol [162, 64, 48, 64] f16.
// ---------------------------------------------------------------------------
__global__ void build_mvol_kernel(const float* __restrict__ feat1,
                                  const float* __restrict__ feat2,
                                  f16* __restrict__ mvol) {
  int idx = blockIdx.x * blockDim.x + threadIdx.x;  // over NB*HW
  if (idx >= NB * HW_FULL) return;
  int nb = idx / HW_FULL, hw = idx - nb * HW_FULL;
  int y = hw >> 6, x = hw & 63;
  int bb = nb / 81, t = nb - bb * 81;
  int iu = t / 9, iv = t - iu * 9;
  int dx = iu - 4, dy = iv - 4;
  int sy = y + dy, sx = x + dx;
  bool ib = (sy >= 0 && sy < 48 && sx >= 0 && sx < 64);
  const float* f1 = feat1 + (size_t)bb * 32 * HW_FULL;
  const float* f2 = feat2 + (size_t)bb * 32 * HW_FULL;
  float sum = 0.f;
  if (ib) {
    for (int c = 0; c < 32; ++c) sum += f2[c * HW_FULL + sy * 64 + sx];
  }
  float valid = (ib && sum != 0.f) ? 1.f : 0.f;
  f16* o = mvol + (size_t)nb * 64 * HW_FULL + hw;
  for (int c = 0; c < 32; ++c) {
    o[c * HW_FULL] = (f16)(f1[c * HW_FULL + hw] * valid);
    float s = ib ? f2[c * HW_FULL + sy * 64 + sx] : 0.f;
    o[(32 + c) * HW_FULL] = (f16)(s * valid);
  }
}

// ---------------------------------------------------------------------------
// Zero-stuffed (lhs_dilation=2) upsample: [162,64,24,32] -> [162,64,47,63].
// ---------------------------------------------------------------------------
__global__ void upsample2x_kernel(const f16* __restrict__ in, f16* __restrict__ out) {
  const long total = (long)NB * 64 * 47 * 63;
  long idx = (long)blockIdx.x * blockDim.x + threadIdx.x;
  if (idx >= total) return;
  int xw = (int)(idx % 63);
  long t = idx / 63;
  int yh = (int)(t % 47);
  t /= 47;
  int ci = (int)(t % 64);
  int n = (int)(t / 64);
  f16 v = (f16)0;
  if (!(yh & 1) && !(xw & 1))
    v = in[(((size_t)n * 64 + ci) * 24 + (yh >> 1)) * 32 + (xw >> 1)];
  out[idx] = v;
}

// ---------------------------------------------------------------------------
// Implicit-GEMM KHxKW conv via V_WMMA_F32_16X16X32_F16.
// Block = one 16-wide spatial tile (tn), image n = blockIdx.y; wave w owns
// Cout tile tm = w. Per 64-wide K round: 256 threads stage the 64x16 im2col
// B tile into LDS (fragment order, division-free incremental tap walk), then
// each compute wave issues 2 WMMAs (A = contiguous packed-weight vectors).
// Wave-uniform `compute` branch keeps EXEC all-ones around the WMMA.
// ---------------------------------------------------------------------------
__global__ __launch_bounds__(256) void conv_wmma_kernel(
    const f16* __restrict__ in, const f16* __restrict__ wpack, f16* __restrict__ out,
    int Cin, int H, int W, int Cout, int stride, int dil, int pad,
    int Ho, int Wo, int KH, int KW, int Kpad) {
  const int taps = KH * KW;
  const int HWo = Ho * Wo;
  const int tn = blockIdx.x, n = blockIdx.y;
  const int tid = threadIdx.x;
  const int tm = tid >> 5;  // wave id = Cout tile
  const int lane = tid & 31;
  const int grp = lane >> 4, lm = lane & 15;
  const int Mt = (Cout + 15) >> 4;
  const bool compute = (tm < Mt);  // wave-uniform
  const f16* inb = in + (size_t)n * Cin * H * W;

  __shared__ __align__(32) f16 lbs[2][32][16];  // two 32xK chunks, fragment order

  // ---- staging state: 4 elements/thread (64 K-rows x 16 cols per round) ----
  const int colT = tid & 15;
  const int ppT = tn * 16 + colT;
  const bool colOK = (ppT < HWo);
  const int oyT = colOK ? ppT / Wo : 0;
  const int oxT = colOK ? ppT - oyT * Wo : 0;
  const int ybase = oyT * stride - pad;
  const int xbase = oxT * stride - pad;
  const int dci = 64 / taps, drem = 64 % taps;
  const int dky = drem / KW, dkx = drem - dky * KW;

  int ciS[4], kyS[4], kxS[4];
  f16* dstS[4];
#pragma unroll
  for (int s = 0; s < 4; ++s) {
    int kk = (tid >> 4) + 16 * s;  // 0..63, fixed per thread
    int ci = kk / taps, r = kk - ci * taps;
    int ky = r / KW, kx = r - ky * KW;
    ciS[s] = ci;
    kyS[s] = ky;
    kxS[s] = kx;
    dstS[s] = &lbs[kk >> 5][((kk & 8) << 1) | colT][(kk & 7) | ((kk & 16) >> 1)];
  }

  const int p = tn * 16 + lm;  // this lane's output column
  const bool pvalid = (p < HWo);
  const f16* arow = wpack + (size_t)(tm * 16 + lm) * Kpad;  // padded: always safe

  v8f c = {};
  for (int k0 = 0; k0 < Kpad; k0 += 64) {
    // ---- stage B (division-free incremental tap walk) ----
#pragma unroll
    for (int s = 0; s < 4; ++s) {
      int iy = ybase + kyS[s] * dil;
      int ix = xbase + kxS[s] * dil;
      f16 v = (f16)0;
      if (colOK && ciS[s] < Cin && iy >= 0 && iy < H && ix >= 0 && ix < W)
        v = inb[((size_t)ciS[s] * H + iy) * W + ix];
      *dstS[s] = v;
      // advance this element's K by 64: r += drem (with carries), ci += dci
      kxS[s] += dkx;
      if (kxS[s] >= KW) { kxS[s] -= KW; kyS[s] += 1; }
      kyS[s] += dky;
      if (kyS[s] >= KH) { kyS[s] -= KH; ciS[s] += 1; }
      ciS[s] += dci;
    }
    __syncthreads();

    if (compute) {
      if (k0 + 64 < Kpad) __builtin_prefetch(arow + k0 + 64, 0, 0);
#pragma unroll
      for (int ch = 0; ch < 2; ++ch) {
        v8h a0 = *(const v8h*)(arow + k0 + ch * 32 + grp * 8);
        v8h a1 = *(const v8h*)(arow + k0 + ch * 32 + 16 + grp * 8);
        v16h a;
#pragma unroll
        for (int i = 0; i < 8; ++i) { a[i] = a0[i]; a[8 + i] = a1[i]; }
        v16h b = *(const v16h*)(&lbs[ch][lane][0]);
        c = __builtin_amdgcn_wmma_f32_16x16x32_f16(false, a, false, b, (short)0, c,
                                                   false, false);
      }
    }
    __syncthreads();
  }

  if (compute) {
    f16* outb = out + (size_t)n * Cout * HWo;
#pragma unroll
    for (int r = 0; r < 8; ++r) {
      int row = tm * 16 + r + grp * 8;
      if (row < Cout && pvalid) outb[(size_t)row * HWo + p] = (f16)c[r];
    }
  }
}

// ---------------------------------------------------------------------------
// BatchNorm (training mode, biased var) stats over N*H*W per channel.
// ---------------------------------------------------------------------------
__global__ __launch_bounds__(256) void bn_stats_kernel(const f16* __restrict__ x,
                                                       float* __restrict__ stats,
                                                       int N, int C, int HW) {
  const int c = blockIdx.x;
  const long total = (long)N * HW;
  float s = 0.f, sq = 0.f;
  for (long i = threadIdx.x; i < total; i += blockDim.x) {
    long n = i / HW, hw = i - n * HW;
    float v = (float)x[((size_t)n * C + c) * HW + hw];
    s += v;
    sq += v * v;
  }
  __shared__ float ss[256], s2[256];
  ss[threadIdx.x] = s;
  s2[threadIdx.x] = sq;
  __syncthreads();
  for (int off = 128; off > 0; off >>= 1) {
    if ((int)threadIdx.x < off) {
      ss[threadIdx.x] += ss[threadIdx.x + off];
      s2[threadIdx.x] += s2[threadIdx.x + off];
    }
    __syncthreads();
  }
  if (threadIdx.x == 0) {
    float m = ss[0] / (float)total;
    stats[c] = m;
    stats[C + c] = s2[0] / (float)total - m * m;
  }
}

__global__ void bn_relu_ip_kernel(f16* __restrict__ x, const float* __restrict__ stats,
                                  const float* __restrict__ g, const float* __restrict__ b,
                                  int C, int HW, long total) {
  long i = (long)blockIdx.x * blockDim.x + threadIdx.x;
  if (i >= total) return;
  int c = (int)((i / HW) % C);
  float m = stats[c], v = stats[C + c];
  float y = ((float)x[i] - m) * rsqrtf(v + 1e-5f) * g[c] + b[c];
  x[i] = (f16)(y > 0.f ? y : 0.f);
}

// ---------------------------------------------------------------------------
// Small direct 3x3 conv (pad=1) with bias, f16 in / f32 out (m5, c6).
// ---------------------------------------------------------------------------
__global__ void conv3x3_direct_kernel(const f16* __restrict__ in, const float* __restrict__ w,
                                      const float* __restrict__ bias, float* __restrict__ out,
                                      int N, int Cin, int H, int W, int Cout) {
  const int HW = H * W;
  long idx = (long)blockIdx.x * blockDim.x + threadIdx.x;
  if (idx >= (long)N * Cout * HW) return;
  int hw = (int)(idx % HW);
  int t = (int)(idx / HW);
  int co = t % Cout, n = t / Cout;
  int oy = hw / W, ox = hw - (hw / W) * W;
  float acc = bias[co];
  const f16* inb = in + (size_t)n * Cin * HW;
  for (int ci = 0; ci < Cin; ++ci)
    for (int ky = 0; ky < 3; ++ky) {
      int iy = oy - 1 + ky;
      if (iy < 0 || iy >= H) continue;
      for (int kx = 0; kx < 3; ++kx) {
        int ix = ox - 1 + kx;
        if (ix < 0 || ix >= W) continue;
        acc += (float)inb[ci * HW + iy * W + ix] * w[((co * Cin + ci) * 3 + ky) * 3 + kx];
      }
    }
  out[idx] = acc;
}

// ---------------------------------------------------------------------------
// DAP: 81x81 1x1 conv over the cost volume.
// ---------------------------------------------------------------------------
__global__ void dap_kernel(const float* __restrict__ w, const float* __restrict__ cost,
                           float* __restrict__ out) {
  int idx = blockIdx.x * blockDim.x + threadIdx.x;  // 2*81*3072
  if (idx >= 2 * 81 * HW_FULL) return;
  int hw = idx % HW_FULL;
  int t = idx / HW_FULL;
  int o = t % 81, b = t / 81;
  const float* cb = cost + (size_t)b * 81 * HW_FULL + hw;
  float acc = 0.f;
  for (int i = 0; i < 81; ++i) acc += w[o * 81 + i] * cb[i * HW_FULL];
  out[idx] = acc;
}

// ---------------------------------------------------------------------------
// Softmax over 81 displacements + soft-argmin flow + normalized entropy.
// ---------------------------------------------------------------------------
__global__ void soft_argmin_kernel(const float* __restrict__ cost, float* __restrict__ flow,
                                   float* __restrict__ entr) {
  int idx = blockIdx.x * blockDim.x + threadIdx.x;  // 2*3072
  if (idx >= 2 * HW_FULL) return;
  int hw = idx % HW_FULL, b = idx / HW_FULL;
  const float* cb = cost + (size_t)b * 81 * HW_FULL + hw;
  float mx = -1e30f;
  for (int i = 0; i < 81; ++i) mx = fmaxf(mx, cb[i * HW_FULL]);
  float sum = 0.f;
  for (int i = 0; i < 81; ++i) sum += __expf(cb[i * HW_FULL] - mx);
  float inv = 1.f / sum;
  float fu = 0.f, fv = 0.f, ent = 0.f;
  for (int i = 0; i < 81; ++i) {
    float p = __expf(cb[i * HW_FULL] - mx) * inv;
    fu += p * (float)(i / 9 - 4);
    fv += p * (float)(i % 9 - 4);
    float pc = fminf(fmaxf(p, 1e-9f), 1.f - 1e-9f);
    ent -= p * __logf(pc);
  }
  flow[(size_t)b * 2 * HW_FULL + hw] = fu;
  flow[(size_t)b * 2 * HW_FULL + HW_FULL + hw] = fv;
  entr[idx] = ent / logf(81.f);
}

// ---------------------------------------------------------------------------
// Bilinear resize (align_corners=True): [2,3,192,256] -> [2,3,48,64].
// ---------------------------------------------------------------------------
__global__ void resize_img_kernel(const float* __restrict__ img, float* __restrict__ out) {
  int idx = blockIdx.x * blockDim.x + threadIdx.x;  // 2*3*3072
  if (idx >= 2 * 3 * HW_FULL) return;
  int hw = idx % HW_FULL;
  int c = (idx / HW_FULL) % 3;
  int b = idx / (3 * HW_FULL);
  int oy = hw >> 6, ox = hw & 63;
  float fy = oy * (191.f / 47.f);
  float fx = ox * (255.f / 63.f);
  int y0 = (int)floorf(fy), x0 = (int)floorf(fx);
  int y1 = min(y0 + 1, 191), x1 = min(x0 + 1, 255);
  float wy = fy - (float)y0, wx = fx - (float)x0;
  const float* p = img + ((size_t)b * 3 + c) * 192 * 256;
  float v00 = p[y0 * 256 + x0], v01 = p[y0 * 256 + x1];
  float v10 = p[y1 * 256 + x0], v11 = p[y1 * 256 + x1];
  out[idx] = (v00 * (1 - wy) + v10 * wy) * (1 - wx) + (v01 * (1 - wy) + v11 * wy) * wx;
}

// ---------------------------------------------------------------------------
// Context-net input assembly: [flow(2), entr(1), feat1(32), img1r(3)] -> f16.
// ---------------------------------------------------------------------------
__global__ void build_ctx_kernel(const float* __restrict__ flow, const float* __restrict__ entr,
                                 const float* __restrict__ feat1, const float* __restrict__ img1r,
                                 f16* __restrict__ ctx) {
  int idx = blockIdx.x * blockDim.x + threadIdx.x;  // 2*3072
  if (idx >= 2 * HW_FULL) return;
  int hw = idx % HW_FULL, b = idx / HW_FULL;
  f16* o = ctx + (size_t)b * 38 * HW_FULL + hw;
  o[0] = (f16)flow[(size_t)b * 2 * HW_FULL + hw];
  o[HW_FULL] = (f16)flow[(size_t)b * 2 * HW_FULL + HW_FULL + hw];
  o[2 * HW_FULL] = (f16)entr[idx];
  for (int c = 0; c < 32; ++c)
    o[(3 + c) * HW_FULL] = (f16)feat1[((size_t)b * 32 + c) * HW_FULL + hw];
  for (int c = 0; c < 3; ++c)
    o[(35 + c) * HW_FULL] = (f16)img1r[((size_t)b * 3 + c) * HW_FULL + hw];
}

__global__ void add_out_kernel(const float* __restrict__ a, const float* __restrict__ b,
                               float* __restrict__ o, int n) {
  int i = blockIdx.x * blockDim.x + threadIdx.x;
  if (i < n) o[i] = a[i] + b[i];
}

// ---------------------------------------------------------------------------
// Host orchestration
// ---------------------------------------------------------------------------
extern "C" void kernel_launch(void* const* d_in, const int* in_sizes, int n_in,
                              void* d_out, int out_size, void* d_ws, size_t ws_size,
                              hipStream_t stream) {
  (void)in_sizes; (void)n_in; (void)out_size; (void)ws_size;
  const float* img1  = (const float*)d_in[0];
  const float* feat1 = (const float*)d_in[1];
  const float* feat2 = (const float*)d_in[2];
  const float* m_w[4] = {(const float*)d_in[3], (const float*)d_in[6],
                         (const float*)d_in[9], (const float*)d_in[12]};
  const float* m_g[4] = {(const float*)d_in[4], (const float*)d_in[7],
                         (const float*)d_in[10], (const float*)d_in[13]};
  const float* m_b[4] = {(const float*)d_in[5], (const float*)d_in[8],
                         (const float*)d_in[11], (const float*)d_in[14]};
  const float* m4_w = (const float*)d_in[15];
  const float* m4_g = (const float*)d_in[16];
  const float* m4_b = (const float*)d_in[17];
  const float* m5_w = (const float*)d_in[18];
  const float* m5_b = (const float*)d_in[19];
  const float* dapw = (const float*)d_in[20];
  const float *c_w[6], *c_g[6], *c_b[6];
  for (int i = 0; i < 6; ++i) {
    c_w[i] = (const float*)d_in[21 + 3 * i];
    c_g[i] = (const float*)d_in[22 + 3 * i];
    c_b[i] = (const float*)d_in[23 + 3 * i];
  }
  const float* c6_w = (const float*)d_in[39];
  const float* c6_b = (const float*)d_in[40];
  float* out = (float*)d_out;

  // ---- workspace carving ----
  char* ws = (char*)d_ws;
  size_t off = 0;
  auto carve = [&](size_t bytes) -> char* {
    char* p = ws + off;
    off += (bytes + 255) & ~(size_t)255;
    return p;
  };
  const size_t actElems = (size_t)NB * 96 * HW_FULL;  // largest activation
  f16* act0 = (f16*)carve(actElems * 2);
  f16* act1 = (f16*)carve(actElems * 2);
  f16* actU = (f16*)carve((size_t)NB * 64 * 47 * 63 * 2);  // zero-stuffed deconv input
  f16* w16  = (f16*)carve(4u << 20);
  float* stats = (float*)carve(4096);
  float* cost  = (float*)carve((size_t)2 * 81 * HW_FULL * 4);
  float* dapo  = (float*)carve((size_t)2 * 81 * HW_FULL * 4);
  float* flow  = (float*)carve((size_t)2 * 2 * HW_FULL * 4);
  float* entr  = (float*)carve((size_t)2 * HW_FULL * 4);
  float* img1r = (float*)carve((size_t)2 * 3 * HW_FULL * 4);
  f16*   ctx   = (f16*)carve((size_t)2 * 38 * HW_FULL * 2);
  float* ybuf  = (float*)carve((size_t)2 * 2 * HW_FULL * 4);

  // ---- weight repack into padded WMMA-friendly layout (3x3 layers) ----
  struct WSpec { const float* src; int Cout; int Cin; };
  const WSpec wspec[10] = {
      {m_w[0], 96, 64},  {m_w[1], 128, 96}, {m_w[2], 128, 128}, {m_w[3], 64, 128},
      {c_w[0], 64, 38},  {c_w[1], 128, 64}, {c_w[2], 128, 128}, {c_w[3], 96, 128},
      {c_w[4], 64, 96},  {c_w[5], 32, 64}};
  f16* wptr[10];
  int kpad[10];
  size_t woff = 0;
  for (int i = 0; i < 10; ++i) {
    int Ktot = wspec[i].Cin * 9;
    int Kpad = (Ktot + 63) & ~63;
    int rows = (wspec[i].Cout + 15) & ~15;
    wptr[i] = w16 + woff;
    kpad[i] = Kpad;
    woff += (size_t)rows * Kpad;
    int n = rows * Kpad;
    pack_wgt_kernel<<<(n + 255) / 256, 256, 0, stream>>>(wspec[i].src, wptr[i],
                                                         wspec[i].Cout, Ktot, Kpad, rows);
  }
  f16* wdec = w16 + woff;  // deconv: 32 rows x 1024 K
  pack_deconv_kernel<<<(32 * 1024 + 255) / 256, 256, 0, stream>>>(m4_w, wdec);

  auto conv = [&](const f16* in, const f16* w, f16* o, int N, int Cin, int H, int W,
                  int Cout, int stride, int dil, int pad, int Ho, int Wo,
                  int KH, int KW, int Kp) {
    dim3 grid((Ho * Wo + 15) / 16, N);
    conv_wmma_kernel<<<grid, 256, 0, stream>>>(in, w, o, Cin, H, W, Cout, stride,
                                               dil, pad, Ho, Wo, KH, KW, Kp);
  };
  auto bn = [&](f16* x, int N, int C, int HW, const float* g, const float* b) {
    bn_stats_kernel<<<C, 256, 0, stream>>>(x, stats, N, C, HW);
    long total = (long)N * C * HW;
    bn_relu_ip_kernel<<<(int)((total + 255) / 256), 256, 0, stream>>>(x, stats, g, b,
                                                                      C, HW, total);
  };

  // ---- cost volume ----
  {
    int n = NB * HW_FULL;
    build_mvol_kernel<<<(n + 255) / 256, 256, 0, stream>>>(feat1, feat2, act0);
  }

  // ---- matching network ----
  conv(act0, wptr[0], act1, NB, 64, 48, 64, 96, 1, 1, 1, 48, 64, 3, 3, kpad[0]);
  bn(act1, NB, 96, HW_FULL, m_g[0], m_b[0]);
  conv(act1, wptr[1], act0, NB, 96, 48, 64, 128, 2, 1, 1, 24, 32, 3, 3, kpad[1]);
  bn(act0, NB, 128, 24 * 32, m_g[1], m_b[1]);
  conv(act0, wptr[2], act1, NB, 128, 24, 32, 128, 1, 1, 1, 24, 32, 3, 3, kpad[2]);
  bn(act1, NB, 128, 24 * 32, m_g[2], m_b[2]);
  conv(act1, wptr[3], act0, NB, 128, 24, 32, 64, 1, 1, 1, 24, 32, 3, 3, kpad[3]);
  bn(act0, NB, 64, 24 * 32, m_g[3], m_b[3]);
  // m4 deconv = zero-stuffed upsample + 4x4 WMMA conv (pad 2)
  {
    long n = (long)NB * 64 * 47 * 63;
    upsample2x_kernel<<<(int)((n + 255) / 256), 256, 0, stream>>>(act0, actU);
  }
  conv(actU, wdec, act1, NB, 64, 47, 63, 32, 1, 1, 2, 48, 64, 4, 4, 1024);
  bn(act1, NB, 32, HW_FULL, m4_g, m4_b);
  {
    long n = (long)NB * 1 * HW_FULL;  // m5 output is already [2,81,48,64] layout
    conv3x3_direct_kernel<<<(int)((n + 255) / 256), 256, 0, stream>>>(
        act1, m5_w, m5_b, cost, NB, 32, 48, 64, 1);
  }

  // ---- DAP + soft-argmin + entropy ----
  {
    int n = 2 * 81 * HW_FULL;
    dap_kernel<<<(n + 255) / 256, 256, 0, stream>>>(dapw, cost, dapo);
  }
  {
    int n = 2 * HW_FULL;
    soft_argmin_kernel<<<(n + 255) / 256, 256, 0, stream>>>(dapo, flow, entr);
  }

  // ---- context network ----
  {
    int n = 2 * 3 * HW_FULL;
    resize_img_kernel<<<(n + 255) / 256, 256, 0, stream>>>(img1, img1r);
    int m = 2 * HW_FULL;
    build_ctx_kernel<<<(m + 255) / 256, 256, 0, stream>>>(flow, entr, feat1, img1r, ctx);
  }
  const int cci[6] = {38, 64, 128, 128, 96, 64};
  const int cco[6] = {64, 128, 128, 96, 64, 32};
  const int cdil[6] = {1, 2, 4, 8, 16, 1};
  f16* cin = ctx;
  f16* cbuf[2] = {act0, act1};
  for (int i = 0; i < 6; ++i) {
    f16* co = cbuf[i & 1];
    conv(cin, wptr[4 + i], co, 2, cci[i], 48, 64, cco[i], 1, cdil[i], cdil[i],
         48, 64, 3, 3, kpad[4 + i]);
    bn(co, 2, cco[i], HW_FULL, c_g[i], c_b[i]);
    cin = co;
  }
  {
    long n = (long)2 * 2 * HW_FULL;
    conv3x3_direct_kernel<<<(int)((n + 255) / 256), 256, 0, stream>>>(
        cin, c6_w, c6_b, ybuf, 2, 32, 48, 64, 2);
    add_out_kernel<<<(int)((n + 255) / 256), 256, 0, stream>>>(flow, ybuf, out, (int)n);
  }
}
